// VanillaSFNN_86474871538194
// MI455X (gfx1250) — compile-verified, matrix operand order checked by reading, not run on hardware
//
#include <hip/hip_runtime.h>

#define HIDDEN  256
#define BATCH   1024
#define TSTEPS  1024
#define IN_DIM  2
#define OUT_DIM 2
#define DECAY   0.8f

#define WAVES_PER_BLOCK 16   // one batch element per wave; 16 rows = WMMA M tile
#define HPL 8                // hidden units per lane (256 / 32)

typedef __attribute__((ext_vector_type(16))) _Float16 v16h;
typedef __attribute__((ext_vector_type(8)))  _Float16 v8h;
typedef __attribute__((ext_vector_type(8)))  float    v8f;

__global__ __launch_bounds__(WAVES_PER_BLOCK * 32, 2)
void sfnn_fused_kernel(const float* __restrict__ x_seq,   // [B,T,2]
                       const float* __restrict__ W_in,    // [256,2]
                       const float* __restrict__ b_in,    // [256]
                       const float* __restrict__ W_out,   // [2,256]
                       const float* __restrict__ b_out,   // [2]
                       const float* __restrict__ tau_m,   // [256]
                       float* __restrict__ out)           // [B,2]
{
    __shared__ _Float16 accH[WAVES_PER_BLOCK][HIDDEN];    // A tile: 16 x 256 f16
    __shared__ _Float16 WoutB[16][HIDDEN];                // B tile: rows 2..15 zero

    const int tid  = threadIdx.x;
    const int wave = tid >> 5;
    const int lane = tid & 31;
    const int b    = blockIdx.x * WAVES_PER_BLOCK + wave;

    // Stage zero-padded B tile: thread t owns halves [t*8, t*8+8) — disjoint, no race.
    {
        const int base = tid * 8;                 // flat f16 index into WoutB
        v8h w = {};
        if (base < OUT_DIM * HIDDEN) {
            #pragma unroll
            for (int m = 0; m < 8; ++m)
                w[m] = (_Float16)W_out[base + m];
        }
        ((v8h*)WoutB)[tid] = w;                   // one ds_store_b128
    }

    // ---- per-lane constants for HPL hidden units: h = lane*HPL + j ----
    float alpha[HPL], c0[HPL], c1[HPL], cb[HPL], v[HPL], acc[HPL];
    #pragma unroll
    for (int j = 0; j < HPL; ++j) {
        const int h = lane * HPL + j;
        const float a     = 1.0f / (1.0f + __expf(-tau_m[h]));   // sigmoid
        const float one_m = 1.0f - a;
        alpha[j] = a;
        c0[j]    = one_m * W_in[h * IN_DIM + 0];
        c1[j]    = one_m * W_in[h * IN_DIM + 1];
        cb[j]    = one_m * b_in[h];
        v[j]     = 0.0f;
        acc[j]   = 0.0f;   // decayed spike accumulator
    }

    // ---- sequential LIF scan over T (VALU-bound core) ----
    // xl: 32 timesteps of this wave's batch row, one per lane; double-buffered.
    const float2* __restrict__ xp = (const float2*)x_seq + (size_t)b * TSTEPS;
    float2 xl = xp[lane];
    for (int tb = 0; tb < TSTEPS; tb += 32) {
        float2 xn = (tb + 32 < TSTEPS) ? xp[tb + 32 + lane] : make_float2(0.f, 0.f);
        #pragma unroll
        for (int jj = 0; jj < 32; ++jj) {
            const float x0 = __shfl(xl.x, jj, 32);
            const float x1 = __shfl(xl.y, jj, 32);
            #pragma unroll
            for (int k = 0; k < HPL; ++k) {
                const float cur = fmaf(c1[k], x1, fmaf(c0[k], x0, cb[k]));
                const float vv  = fmaf(alpha[k], v[k], cur);     // membrane update
                const float sp  = (vv >= 1.0f) ? 1.0f : 0.0f;    // threshold
                v[k]   = vv - sp;                                // soft reset
                acc[k] = fmaf(acc[k], DECAY, sp);                // decayed spike sum
            }
        }
        xl = xn;
    }

    // ---- stash acc as row `wave` of the f16 A tile (one 16B DS store) ----
    {
        v8h apk;
        #pragma unroll
        for (int j = 0; j < HPL; ++j)
            apk[j] = (_Float16)acc[j];
        *(v8h*)&accH[wave][lane * HPL] = apk;
    }
    __syncthreads();

    // ---- wave 0: integ[16x2] = accTile[16x256] @ W_out^T, 8 chained WMMAs ----
    if (wave == 0) {                       // wave-uniform branch: EXEC all ones
        v8f c = {};
        const int rc  = lane & 15;               // A row / B column
        const int khi = (lane & 16) ? 8 : 0;     // high-half K offset per ISA layout
        const _Float16* __restrict__ arow = &accH[rc][0];
        const _Float16* __restrict__ brow = &WoutB[rc][0];
        #pragma unroll
        for (int k0 = 0; k0 < HIDDEN; k0 += 32) {
            // 16-bit A 16x32 layout: lanes<16 hold K 0..7 & 16..23, lanes>=16 K 8..15 & 24..31
            const v8h alo = *(const v8h*)(arow + k0 + khi);        // ds_load_b128
            const v8h ahi = *(const v8h*)(arow + k0 + 16 + khi);
            const v8h blo = *(const v8h*)(brow + k0 + khi);
            const v8h bhi = *(const v8h*)(brow + k0 + 16 + khi);
            const v16h a  = __builtin_shufflevector(alo, ahi,
                                0,1,2,3,4,5,6,7,8,9,10,11,12,13,14,15);
            const v16h bm = __builtin_shufflevector(blo, bhi,
                                0,1,2,3,4,5,6,7,8,9,10,11,12,13,14,15);
            c = __builtin_amdgcn_wmma_f32_16x16x32_f16(false, a, false, bm,
                                                       (short)0, c, false, false);
        }
        // bias contribution: b_out * sum_{k=0}^{T-1} DECAY^k
        const float geo = (1.0f - __powf(DECAY, (float)TSTEPS)) / (1.0f - DECAY);
        const int n   = lane & 15;          // output column
        const int mhi = (lane >> 4) * 8;    // C layout: lanes>=16 hold M=8..15
        if (n < OUT_DIM) {
            const float bo = b_out[n] * geo;
            #pragma unroll
            for (int r = 0; r < 8; ++r) {
                const int bb = blockIdx.x * WAVES_PER_BLOCK + mhi + r;
                out[bb * OUT_DIM + n] = c[r] + bo;
            }
        }
    }
}

extern "C" void kernel_launch(void* const* d_in, const int* in_sizes, int n_in,
                              void* d_out, int out_size, void* d_ws, size_t ws_size,
                              hipStream_t stream) {
    const float* x_seq = (const float*)d_in[0];
    const float* W_in  = (const float*)d_in[1];
    const float* b_in  = (const float*)d_in[2];
    const float* W_out = (const float*)d_in[3];
    const float* b_out = (const float*)d_in[4];
    const float* tau_m = (const float*)d_in[5];
    float* out = (float*)d_out;

    dim3 grid(BATCH / WAVES_PER_BLOCK);       // 64 blocks
    dim3 block(WAVES_PER_BLOCK * 32);         // 512 threads = 16 waves
    hipLaunchKernelGGL(sfnn_fused_kernel, grid, block, 0, stream,
                       x_seq, W_in, b_in, W_out, b_out, tau_m, out);
}
